// Seq2Seq_64037962384037
// MI455X (gfx1250) — matrix operand order repaired and checked by reading, not compiled, not run on hardware
//
#include <hip/hip_runtime.h>

typedef __attribute__((ext_vector_type(16))) __bf16 v16bf;
typedef __attribute__((ext_vector_type(8)))  __bf16 v8bf;
typedef __attribute__((ext_vector_type(8)))  float  v8f;

namespace {
constexpr int kB = 64, kS = 512, kF = 64, kEE = 256, kH = 512, kDH = 512,
              kDE = 256, kO = 64, kT = 24;
constexpr int kH3 = 3 * kH;    // 1536
constexpr int kBS = kB * kS;   // 32768
constexpr int kE2 = 2 * kH;    // 1024 (enc_out feature dim)
constexpr int SCAN_WGS = 64;   // workgroups per direction in persistent scan
}

__device__ __forceinline__ float sigm(float x) { return 1.f / (1.f + __expf(-x)); }

// Merge two aligned 8x-bf16 chunks into one WMMA A/B fragment.
__device__ __forceinline__ v16bf load_frag16(const __bf16* p0, const __bf16* p1) {
  const v8bf lo = *(const v8bf*)p0;
  const v8bf hi = *(const v8bf*)p1;
  return __builtin_shufflevector(lo, hi, 0, 1, 2, 3, 4, 5, 6, 7, 8, 9, 10, 11,
                                 12, 13, 14, 15);
}

// ---------------------------------------------------------------------------
// Generic WMMA GEMM: C[M,N] = act(A[M,K] @ op(W) + bias), bf16 MACs, f32 acc.
//   ABF : A stored as bf16 (else f32, converted on load)
//   WNT : weight stored (N,K) row-major (i.e. computing A @ W^T); else (K,N)
//   EPI : 0 = f32 out, 1 = tanh f32 out, 2 = bf16 out
// W is ALWAYS pre-converted bf16. Block = 256 threads = 8 waves ->
// 64(M) x 32(N) tile; grid = (N/32, M/64).
// ---------------------------------------------------------------------------
enum { EPI_F32 = 0, EPI_TANH = 1, EPI_BF16 = 2 };

template <bool ABF, bool WNT, int EPI>
__global__ void wmma_gemm_kernel(const void* __restrict__ Ap, int lda,
                                 const __bf16* __restrict__ W, int ldw,
                                 const float* __restrict__ bias,
                                 void* __restrict__ Cp, int ldc, int K) {
  const int lane = threadIdx.x & 31;
  const int wave = threadIdx.x >> 5;
  const int half = lane >> 4;
  const int lm = lane & 15;
  const int m0 = blockIdx.y * 64 + (wave & 3) * 16;
  const int n0 = blockIdx.x * 32 + (wave >> 2) * 16;
  const int row = m0 + lm;
  const int col = n0 + lm;
  v8f acc = {};
  for (int k0 = 0; k0 < K; k0 += 32) {
    v16bf a, b;
    if constexpr (ABF) {
      const __bf16* ar = (const __bf16*)Ap + (size_t)row * lda + k0 + 16 * half;
      a = load_frag16(ar, ar + 8);
    } else {
      const float* ar = (const float*)Ap + (size_t)row * lda;
#pragma unroll
      for (int e = 0; e < 8; ++e) {
        a[e] = (__bf16)ar[k0 + 16 * half + e];
        a[8 + e] = (__bf16)ar[k0 + 16 * half + 8 + e];
      }
    }
    if constexpr (WNT) {
      const __bf16* wr = W + (size_t)col * ldw + k0 + 16 * half;
      b = *(const v16bf*)wr;  // 32B aligned contiguous run
    } else {
      const __bf16* wc = W + (size_t)(k0 + 16 * half) * ldw + col;
#pragma unroll
      for (int e = 0; e < 16; ++e) b[e] = wc[(size_t)e * ldw];
    }
    acc = __builtin_amdgcn_wmma_f32_16x16x32_bf16(false, a, false, b, (short)0,
                                                  acc, false, false);
  }
  const float bv = bias ? bias[col] : 0.f;
#pragma unroll
  for (int r = 0; r < 8; ++r) {
    float v = acc[r] + bv;
    if constexpr (EPI == EPI_TANH) v = tanhf(v);
    const size_t idx = (size_t)(m0 + 8 * half + r) * ldc + col;
    if constexpr (EPI == EPI_BF16)
      ((__bf16*)Cp)[idx] = (__bf16)v;
    else
      ((float*)Cp)[idx] = v;
  }
}

// One-shot f32 -> bf16 weight conversion.
__global__ void f32_to_bf16_kernel(const float* __restrict__ src,
                                   __bf16* __restrict__ dst, int n) {
  const int i = blockIdx.x * blockDim.x + threadIdx.x;
  if (i < n) dst[i] = (__bf16)src[i];
}

// ---------------------------------------------------------------------------
// Device-wide sense-reversing barrier (per scan direction).
// ---------------------------------------------------------------------------
__device__ __forceinline__ void grid_bar(unsigned* cnt, unsigned* sense,
                                         unsigned nwg, unsigned& gen) {
  __syncthreads();
  if (threadIdx.x == 0) {
    ++gen;
    __threadfence();
    unsigned prev = atomicAdd(cnt, 1u);
    if (prev == nwg - 1u) {
      atomicExch(cnt, 0u);
      __threadfence();
      atomicAdd(sense, 1u);
    } else {
      while (atomicAdd(sense, 0u) < gen) __builtin_amdgcn_s_sleep(1);
      __threadfence();
    }
  }
  __syncthreads();
}

__global__ void init_bar_kernel(unsigned* bar) {
  if (threadIdx.x < 8) bar[threadIdx.x] = 0u;
}

// ---------------------------------------------------------------------------
// Persistent bidirectional GRU scan. grid = (SCAN_WGS, 2), block = 256.
// Per step: (1) gi = emb_s @ Wih^T, gh = h @ Whh^T via WMMA (768 tiles over
// 512 waves), barrier, (2) GRU cell elementwise -> h (f32 + bf16 shadow),
// enc_out slice, barrier. Weights are pre-converted bf16.
// ---------------------------------------------------------------------------
__global__ void gru_scan_kernel(
    const __bf16* __restrict__ emb,  // (B,S,EE) bf16
    const __bf16* __restrict__ Wih_f, const __bf16* __restrict__ Whh_f,
    const float* __restrict__ bih_f, const float* __restrict__ bhh_f,
    const __bf16* __restrict__ Wih_b, const __bf16* __restrict__ Whh_b,
    const float* __restrict__ bih_b, const float* __restrict__ bhh_b,
    float* __restrict__ h_f, float* __restrict__ h_b,    // (B,H) f32 outputs
    __bf16* __restrict__ hbf_f, __bf16* __restrict__ hbf_b,  // bf16 shadows
    float* __restrict__ gi_all, float* __restrict__ gh_all,  // 2 x (B,3H)
    __bf16* __restrict__ enc_out,  // (B,S,2H) bf16
    unsigned* __restrict__ bar) {
  const int dir = blockIdx.y;
  const __bf16* Wih = dir ? Wih_b : Wih_f;
  const __bf16* Whh = dir ? Whh_b : Whh_f;
  const float* bi = dir ? bih_b : bih_f;
  const float* bh = dir ? bhh_b : bhh_f;
  float* h = dir ? h_b : h_f;
  __bf16* hbf = dir ? hbf_b : hbf_f;
  float* gi = gi_all + (size_t)dir * kB * kH3;
  float* gh = gh_all + (size_t)dir * kB * kH3;
  unsigned* cnt = bar + dir * 2;
  unsigned* sense = bar + dir * 2 + 1;
  unsigned gen = 0;

  // h0 = 0
  for (int i = blockIdx.x * blockDim.x + threadIdx.x; i < kB * kH;
       i += SCAN_WGS * 256) {
    h[i] = 0.f;
    hbf[i] = (__bf16)0.f;
  }
  grid_bar(cnt, sense, SCAN_WGS, gen);

  const int lane = threadIdx.x & 31;
  const int wave = threadIdx.x >> 5;
  const int gwave = blockIdx.x * 8 + wave;  // 0..511
  const int half = lane >> 4;
  const int lm = lane & 15;

  for (int t = 0; t < kS; ++t) {
    const int s = dir ? (kS - 1 - t) : t;
    // ---- GEMM phase: tiles 0..383 -> gi (K=EE), 384..767 -> gh (K=H) ----
    for (int tile = gwave; tile < 768; tile += SCAN_WGS * 8) {
      const bool is_gh = tile >= 384;
      const int id = is_gh ? tile - 384 : tile;
      const int m0 = (id & 3) * 16;
      const int n0 = (id >> 2) * 16;
      const __bf16* W = is_gh ? Whh : Wih;  // (1536, K) row-major (NT)
      const __bf16* Abase =
          is_gh ? (hbf + (size_t)(m0 + lm) * kH)
                : (emb + ((size_t)(m0 + lm) * kS + s) * kEE);
      const int K = is_gh ? kH : kEE;
      const int col = n0 + lm;
      v8f acc = {};
      for (int k0 = 0; k0 < K; k0 += 32) {
        const __bf16* ar = Abase + k0 + 16 * half;
        const v16bf a = load_frag16(ar, ar + 8);
        const v16bf b = *(const v16bf*)(W + (size_t)col * K + k0 + 16 * half);
        acc = __builtin_amdgcn_wmma_f32_16x16x32_bf16(false, a, false, b,
                                                      (short)0, acc, false,
                                                      false);
      }
      float* C = is_gh ? gh : gi;
#pragma unroll
      for (int r = 0; r < 8; ++r)
        C[(size_t)(m0 + 8 * half + r) * kH3 + col] = acc[r];
    }
    grid_bar(cnt, sense, SCAN_WGS, gen);

    // ---- GRU cell phase ----
    for (int i = blockIdx.x * blockDim.x + threadIdx.x; i < kB * kH;
         i += SCAN_WGS * 256) {
      const int b = i >> 9;
      const int k = i & (kH - 1);
      const size_t base = (size_t)b * kH3;
      const float r = sigm(gi[base + k] + bi[k] + gh[base + k] + bh[k]);
      const float z =
          sigm(gi[base + kH + k] + bi[kH + k] + gh[base + kH + k] + bh[kH + k]);
      const float n = tanhf(gi[base + 2 * kH + k] + bi[2 * kH + k] +
                            r * (gh[base + 2 * kH + k] + bh[2 * kH + k]));
      const float h2 = (1.f - z) * n + z * h[i];
      h[i] = h2;
      hbf[i] = (__bf16)h2;
      enc_out[((size_t)b * kS + s) * kE2 + dir * kH + k] = (__bf16)h2;
    }
    grid_bar(cnt, sense, SCAN_WGS, gen);
  }
}

// ---------------------------------------------------------------------------
// Small helper kernels
// ---------------------------------------------------------------------------
__global__ void pack_hcat_kernel(const float* __restrict__ h_f,
                                 const float* __restrict__ h_b,
                                 float* __restrict__ hcat) {  // (B, 2H)
  const int i = blockIdx.x * blockDim.x + threadIdx.x;  // B*1024 = 65536
  const int b = i >> 10, k = i & 1023;
  hcat[i] = (k < kH) ? h_f[b * kH + k] : h_b[b * kH + (k - kH)];
}

// score[b,s] = sum_d tanh(encA[b,s,d] + hidA[b,d]) * v_w[d]; one wave per row
__global__ void attn_score_kernel(const __bf16* __restrict__ encA,
                                  const float* __restrict__ hidA,
                                  const float* __restrict__ v_w,
                                  float* __restrict__ score) {
  const int gw = (blockIdx.x * blockDim.x + threadIdx.x) >> 5;  // 0..B*S-1
  const int lane = threadIdx.x & 31;
  const int b = gw >> 9;
  const __bf16* e = encA + (size_t)gw * kDH;
  const float* ha = hidA + (size_t)b * kDH;
  float acc = 0.f;
  for (int d = lane; d < kDH; d += 32)
    acc += tanhf((float)e[d] + ha[d]) * v_w[d];
#pragma unroll
  for (int off = 16; off > 0; off >>= 1) acc += __shfl_xor(acc, off, 32);
  if (lane == 0) score[gw] = acc;
}

// per-batch softmax over S, then ctx[b,:] = sum_s aw[s]*enc_out[b,s,:]
// writes ctx into rnn_in[:,256:1280] and out_cat[:,512:1536]
__global__ void softmax_ctx_kernel(const float* __restrict__ score,
                                   const __bf16* __restrict__ enc_out,
                                   float* __restrict__ rnn_in,
                                   float* __restrict__ out_cat) {
  __shared__ float aw[kS];
  __shared__ float red[256];
  const int b = blockIdx.x;
  const int tid = threadIdx.x;
  const float* sc = score + (size_t)b * kS;
  float mx = -1e30f;
  for (int i = tid; i < kS; i += 256) mx = fmaxf(mx, sc[i]);
  red[tid] = mx;
  __syncthreads();
  for (int w = 128; w > 0; w >>= 1) {
    if (tid < w) red[tid] = fmaxf(red[tid], red[tid + w]);
    __syncthreads();
  }
  mx = red[0];
  __syncthreads();
  float sum = 0.f;
  for (int i = tid; i < kS; i += 256) {
    const float ev = __expf(sc[i] - mx);
    aw[i] = ev;
    sum += ev;
  }
  red[tid] = sum;
  __syncthreads();
  for (int w = 128; w > 0; w >>= 1) {
    if (tid < w) red[tid] += red[tid + w];
    __syncthreads();
  }
  const float inv = 1.f / red[0];
  __syncthreads();
  for (int e = tid; e < kE2; e += 256) {
    const __bf16* col = enc_out + (size_t)b * kS * kE2 + e;
    float acc = 0.f;
    for (int s = 0; s < kS; ++s) acc += aw[s] * (float)col[(size_t)s * kE2];
    acc *= inv;
    rnn_in[(size_t)b * (kDE + kE2) + kDE + e] = acc;
    out_cat[(size_t)b * 1792 + kDH + e] = acc;
  }
}

// copy decoder embedding e (rnn_in[:,0:256]) into out_cat[:,1536:1792]
__global__ void pack_e_kernel(const float* __restrict__ rnn_in,
                              float* __restrict__ out_cat) {
  const int i = blockIdx.x * blockDim.x + threadIdx.x;  // B*256
  const int b = i >> 8, k = i & 255;
  out_cat[(size_t)b * 1792 + 1536 + k] = rnn_in[(size_t)b * 1280 + k];
}

// decoder GRU cell: updates hid in place, mirrors h2 into out_cat[:,0:512]
__global__ void dec_cell_kernel(const float* __restrict__ gi,
                                const float* __restrict__ gh,
                                const float* __restrict__ bi,
                                const float* __restrict__ bh,
                                float* __restrict__ hid,
                                float* __restrict__ out_cat) {
  const int i = blockIdx.x * blockDim.x + threadIdx.x;  // B*DH
  const int b = i >> 9, k = i & (kDH - 1);
  const size_t base = (size_t)b * kH3;
  const float r = sigm(gi[base + k] + bi[k] + gh[base + k] + bh[k]);
  const float z =
      sigm(gi[base + kDH + k] + bi[kDH + k] + gh[base + kDH + k] + bh[kDH + k]);
  const float n = tanhf(gi[base + 2 * kDH + k] + bi[2 * kDH + k] +
                        r * (gh[base + 2 * kDH + k] + bh[2 * kDH + k]));
  const float h2 = (1.f - z) * n + z * hid[i];
  hid[i] = h2;
  out_cat[(size_t)b * 1792 + k] = h2;
}

__global__ void copy_pred_kernel(const float* __restrict__ pred,
                                 float* __restrict__ out, int t) {
  const int i = blockIdx.x * blockDim.x + threadIdx.x;  // B*O
  const int b = i >> 6, o = i & 63;
  out[((size_t)b * kT + t) * kO + o] = pred[i];
}

// ---------------------------------------------------------------------------
extern "C" void kernel_launch(void* const* d_in, const int* in_sizes, int n_in,
                              void* d_out, int out_size, void* d_ws,
                              size_t ws_size, hipStream_t stream) {
  (void)in_sizes; (void)n_in; (void)out_size; (void)ws_size;
  const float* x      = (const float*)d_in[0];
  const float* cast_w = (const float*)d_in[1];
  const float* cast_b = (const float*)d_in[2];
  const float* emb_w  = (const float*)d_in[3];
  const float* emb_b  = (const float*)d_in[4];
  const float* Wih_f  = (const float*)d_in[5];
  const float* Whh_f  = (const float*)d_in[6];
  const float* bih_f  = (const float*)d_in[7];
  const float* bhh_f  = (const float*)d_in[8];
  const float* Wih_b  = (const float*)d_in[9];
  const float* Whh_b  = (const float*)d_in[10];
  const float* bih_b  = (const float*)d_in[11];
  const float* bhh_b  = (const float*)d_in[12];
  const float* fc_w   = (const float*)d_in[13];
  const float* fc_b   = (const float*)d_in[14];
  const float* attn_w = (const float*)d_in[15];
  const float* attn_b = (const float*)d_in[16];
  const float* v_w    = (const float*)d_in[17];
  const float* demb_w = (const float*)d_in[18];
  const float* demb_b = (const float*)d_in[19];
  const float* dWih   = (const float*)d_in[20];
  const float* dWhh   = (const float*)d_in[21];
  const float* dbih   = (const float*)d_in[22];
  const float* dbhh   = (const float*)d_in[23];
  const float* out_w  = (const float*)d_in[24];
  const float* out_b  = (const float*)d_in[25];
  float* out = (float*)d_out;

  // ---- workspace carve-up ----
  char* base = (char*)d_ws;
  size_t off = 0;
  auto alloc = [&](size_t bytes) {
    off = (off + 255) & ~(size_t)255;
    char* p = base + off;
    off += bytes;
    return p;
  };
  __bf16* emb_bf  = (__bf16*)alloc((size_t)kBS * kEE * 2);  // 16 MB
  __bf16* enc_out = (__bf16*)alloc((size_t)kBS * kE2 * 2);  // 64 MB
  __bf16* encA    = (__bf16*)alloc((size_t)kBS * kDH * 2);  // 32 MB
  float*  h_f     = (float*)alloc((size_t)kB * kH * 4);
  float*  h_b     = (float*)alloc((size_t)kB * kH * 4);
  __bf16* hbf_f   = (__bf16*)alloc((size_t)kB * kH * 2);
  __bf16* hbf_b   = (__bf16*)alloc((size_t)kB * kH * 2);
  float*  gi_sc   = (float*)alloc((size_t)2 * kB * kH3 * 4);
  float*  gh_sc   = (float*)alloc((size_t)2 * kB * kH3 * 4);
  unsigned* bar   = (unsigned*)alloc(256);
  float*  hcat    = (float*)alloc((size_t)kB * kE2 * 4);
  float*  hid     = (float*)alloc((size_t)kB * kDH * 4);
  float*  cur_in  = (float*)alloc((size_t)kB * kO * 4);
  float*  rnn_in  = (float*)alloc((size_t)kB * 1280 * 4);
  float*  hidA    = (float*)alloc((size_t)kB * kDH * 4);
  float*  score   = (float*)alloc((size_t)kBS * 4);
  float*  gi_d    = (float*)alloc((size_t)kB * kH3 * 4);
  float*  gh_d    = (float*)alloc((size_t)kB * kH3 * 4);
  float*  out_cat = (float*)alloc((size_t)kB * 1792 * 4);

  // bf16 weight arena
  auto walloc = [&](size_t elems) { return (__bf16*)alloc(elems * 2); };
  __bf16* wb_emb  = walloc((size_t)kF * kEE);
  __bf16* wb_cast = walloc((size_t)kF * kO);
  __bf16* wb_ihf  = walloc((size_t)kH3 * kEE);
  __bf16* wb_hhf  = walloc((size_t)kH3 * kH);
  __bf16* wb_ihb  = walloc((size_t)kH3 * kEE);
  __bf16* wb_hhb  = walloc((size_t)kH3 * kH);
  __bf16* wb_fc   = walloc((size_t)kE2 * kDH);
  __bf16* wb_attn = walloc((size_t)(kE2 + kDH) * kDH);
  __bf16* wb_demb = walloc((size_t)kO * kDE);
  __bf16* wb_dih  = walloc((size_t)kH3 * 1280);
  __bf16* wb_dhh  = walloc((size_t)kH3 * kDH);
  __bf16* wb_out  = walloc((size_t)1792 * kO);

  auto cvt = [&](const float* s, __bf16* d, size_t n) {
    f32_to_bf16_kernel<<<dim3((unsigned)((n + 255) / 256)), 256, 0, stream>>>(
        s, d, (int)n);
  };
  cvt(emb_w,  wb_emb,  (size_t)kF * kEE);
  cvt(cast_w, wb_cast, (size_t)kF * kO);
  cvt(Wih_f,  wb_ihf,  (size_t)kH3 * kEE);
  cvt(Whh_f,  wb_hhf,  (size_t)kH3 * kH);
  cvt(Wih_b,  wb_ihb,  (size_t)kH3 * kEE);
  cvt(Whh_b,  wb_hhb,  (size_t)kH3 * kH);
  cvt(fc_w,   wb_fc,   (size_t)kE2 * kDH);
  cvt(attn_w, wb_attn, (size_t)(kE2 + kDH) * kDH);
  cvt(demb_w, wb_demb, (size_t)kO * kDE);
  cvt(dWih,   wb_dih,  (size_t)kH3 * 1280);
  cvt(dWhh,   wb_dhh,  (size_t)kH3 * kDH);
  cvt(out_w,  wb_out,  (size_t)1792 * kO);

  // 1) emb = bf16(x @ emb_w + emb_b): M=BS, N=EE, K=F, NN weight
  wmma_gemm_kernel<false, false, EPI_BF16>
      <<<dim3(kEE / 32, kBS / 64), 256, 0, stream>>>(x, kF, wb_emb, kEE, emb_b,
                                                     emb_bf, kEE, kF);
  // 2) persistent bidirectional GRU scan
  init_bar_kernel<<<1, 32, 0, stream>>>(bar);
  gru_scan_kernel<<<dim3(SCAN_WGS, 2), 256, 0, stream>>>(
      emb_bf, wb_ihf, wb_hhf, bih_f, bhh_f, wb_ihb, wb_hhb, bih_b, bhh_b, h_f,
      h_b, hbf_f, hbf_b, gi_sc, gh_sc, enc_out, bar);
  // 3) hidden = tanh([h_f|h_b] @ fc_w + fc_b): M=64, N=DH, K=2H
  pack_hcat_kernel<<<256, 256, 0, stream>>>(h_f, h_b, hcat);
  wmma_gemm_kernel<false, false, EPI_TANH>
      <<<dim3(kDH / 32, 1), 256, 0, stream>>>(hcat, kE2, wb_fc, kDH, fc_b, hid,
                                              kDH, kE2);
  // 4) hoisted attention term: encA = bf16(enc_out @ attn_w[:2H] + attn_b)
  wmma_gemm_kernel<true, false, EPI_BF16>
      <<<dim3(kDH / 32, kBS / 64), 256, 0, stream>>>(enc_out, kE2, wb_attn,
                                                     kDH, attn_b, encA, kDH,
                                                     kE2);
  // 5) cur_in = x[:,-1,:] @ cast_w + cast_b   (strided A, lda = S*F)
  wmma_gemm_kernel<false, false, EPI_F32>
      <<<dim3(kO / 32, 1), 256, 0, stream>>>(x + (size_t)(kS - 1) * kF,
                                             kS * kF, wb_cast, kO, cast_b,
                                             cur_in, kO, kF);
  // 6) decoder loop
  for (int t = 0; t < kT; ++t) {
    // e = cur_in @ demb_w + demb_b  -> rnn_in[:, 0:256]
    wmma_gemm_kernel<false, false, EPI_F32>
        <<<dim3(kDE / 32, 1), 256, 0, stream>>>(cur_in, kO, wb_demb, kDE,
                                                demb_b, rnn_in, 1280, kO);
    pack_e_kernel<<<64, 256, 0, stream>>>(rnn_in, out_cat);
    // hidA = hid @ attn_w[2H:, :]
    wmma_gemm_kernel<false, false, EPI_F32>
        <<<dim3(kDH / 32, 1), 256, 0, stream>>>(hid, kDH,
                                                wb_attn + (size_t)kE2 * kDH,
                                                kDH, nullptr, hidA, kDH, kDH);
    attn_score_kernel<<<4096, 256, 0, stream>>>(encA, hidA, v_w, score);
    softmax_ctx_kernel<<<kB, 256, 0, stream>>>(score, enc_out, rnn_in, out_cat);
    // gi = rnn_in @ dWih^T (NT), gh = hid @ dWhh^T (NT)
    wmma_gemm_kernel<false, true, EPI_F32>
        <<<dim3(kH3 / 32, 1), 256, 0, stream>>>(rnn_in, 1280, wb_dih, 1280,
                                                nullptr, gi_d, kH3, 1280);
    wmma_gemm_kernel<false, true, EPI_F32>
        <<<dim3(kH3 / 32, 1), 256, 0, stream>>>(hid, kDH, wb_dhh, kDH, nullptr,
                                                gh_d, kH3, kDH);
    dec_cell_kernel<<<128, 256, 0, stream>>>(gi_d, gh_d, dbih, dbhh, hid,
                                             out_cat);
    // pred = out_cat @ out_w + out_b -> cur_in (feeds next step)
    wmma_gemm_kernel<false, false, EPI_F32>
        <<<dim3(kO / 32, 1), 256, 0, stream>>>(out_cat, 1792, wb_out, kO,
                                               out_b, cur_in, kO, 1792);
    copy_pred_kernel<<<16, 256, 0, stream>>>(cur_in, out, t);
  }
}
